// Model2_60833916780915
// MI455X (gfx1250) — compile-verified, matrix operand order checked by reading, not run on hardware
//
#include <hip/hip_runtime.h>
#include <hip/hip_bf16.h>

// ---------------- types / constants ----------------
typedef unsigned short ushortT;
typedef __attribute__((ext_vector_type(16))) __bf16 v16bf;
typedef __attribute__((ext_vector_type(8)))  float  v8f;
typedef __attribute__((ext_vector_type(4)))  unsigned int u32x4;

#define GH      512          // hidden H
#define TWO_H   1024
#define THREE_H 1536
#define NDOC    40
#define SPD     25
#define LW      40
#define EPD     10
#define EVL     4
#define DEMB    300
#define DPAD    320
#define PDOC    20
#define PSENT   30
#define PDIM    50

#define WMMA_BF16(A_, B_, C_) \
  __builtin_amdgcn_wmma_f32_16x16x32_bf16(false, (A_), false, (B_), (short)0, (C_), false, false)

// ---------------- small helpers ----------------
__device__ __forceinline__ float bf2f(ushortT u) {
  unsigned x = ((unsigned)u) << 16; float f; __builtin_memcpy(&f, &x, 4); return f;
}
__device__ __forceinline__ ushortT f2bf(float f) {
  unsigned x; __builtin_memcpy(&x, &f, 4);
  x = x + 0x7FFFu + ((x >> 16) & 1u);            // round-to-nearest-even
  return (ushortT)(x >> 16);
}
__device__ __forceinline__ float sigmoidf_(float x) { return 1.0f / (1.0f + __expf(-x)); }

// A/B fragment loader per CDNA5 ISA 16-bit A-matrix 16x32 layout:
// lane L in [0,16): K chunks {k0..k0+7, k0+16..k0+23}; lanes 16..31: +8.
__device__ __forceinline__ v16bf load_frag(const ushortT* base, size_t row, int ld, int k0, int lane) {
  int kh = (lane & 16) >> 1;                     // 0 or 8
  const ushortT* p = base + row * (size_t)ld + (size_t)(k0 + kh);
  union { u32x4 u[2]; v16bf v; } t;
  t.u[0] = *(const u32x4*)(p);
  t.u[1] = *(const u32x4*)(p + 16);
  return t.v;
}

// ---------------- conversion / gather kernels ----------------
// dst[r, c] (bf16, [rows, colsD]) = c < colsS ? src[r, c] : 0
__global__ void k_cvt_pad(ushortT* __restrict__ dst, const float* __restrict__ src,
                          int rows, int colsS, int colsD) {
  long idx = (long)blockIdx.x * blockDim.x + threadIdx.x;
  long tot = (long)rows * colsD;
  if (idx >= tot) return;
  int r = (int)(idx / colsD), c = (int)(idx % colsD);
  dst[idx] = (c < colsS) ? f2bf(src[(size_t)r * colsS + c]) : (ushortT)0;
}

// dst[c, r] (bf16, [cols, rows]) = src[r, c]
__global__ void k_cvt_T(ushortT* __restrict__ dst, const float* __restrict__ src, int rows, int cols) {
  long idx = (long)blockIdx.x * blockDim.x + threadIdx.x;
  long tot = (long)rows * cols;
  if (idx >= tot) return;
  int r = (int)(idx / cols), c = (int)(idx % cols);
  dst[(size_t)c * rows + r] = f2bf(src[idx]);
}

// Wcat[r, :] = [ wih[r, 0:dinSrc], zeros(dinPad-dinSrc), whh[r, 0:512] ]
__global__ void k_build_wcat(ushortT* __restrict__ dst, const float* __restrict__ wih,
                             const float* __restrict__ whh, int dinSrc, int dinPad) {
  int ldc = dinPad + GH;
  long idx = (long)blockIdx.x * blockDim.x + threadIdx.x;
  long tot = (long)THREE_H * ldc;
  if (idx >= tot) return;
  int r = (int)(idx / ldc), c = (int)(idx % ldc);
  float v;
  if (c < dinSrc)       v = wih[(size_t)r * dinSrc + c];
  else if (c < dinPad)  v = 0.0f;
  else                  v = whh[(size_t)r * GH + (c - dinPad)];
  dst[idx] = f2bf(v);
}

// gather + pad embeddings: dst [nTok, DPAD] bf16
__global__ void k_gather(ushortT* __restrict__ dst, const int* __restrict__ tok,
                         const float* __restrict__ embed, long nTok) {
  long idx = (long)blockIdx.x * blockDim.x + threadIdx.x;
  long tot = nTok * DPAD;
  if (idx >= tot) return;
  long t = idx / DPAD; int c = (int)(idx % DPAD);
  dst[idx] = (c < DEMB) ? f2bf(embed[(size_t)tok[t] * DEMB + c]) : (ushortT)0;
}

// dst bf16 [rows, ca+cb] = [A | B]; A optionally row-broadcast
__global__ void k_concat_bf16(ushortT* __restrict__ dst, const float* __restrict__ A,
                              const float* __restrict__ B, int rows, int ca, int cb, int bcastA) {
  int ld = ca + cb;
  long idx = (long)blockIdx.x * blockDim.x + threadIdx.x;
  long tot = (long)rows * ld;
  if (idx >= tot) return;
  int r = (int)(idx / ld), c = (int)(idx % ld);
  float v = (c < ca) ? A[(size_t)(bcastA ? 0 : r) * ca + c]
                     : B[(size_t)r * cb + (c - ca)];
  dst[idx] = f2bf(v);
}

// ---------------- WMMA GEMM: C[M,N] = A[M,K] @ W[N,K]^T (+bias) ----------------
// one wave per 16x64 tile; N % 64 == 0, K % 32 == 0
__global__ __launch_bounds__(32) void k_gemm_bf16(
    const ushortT* __restrict__ A, const ushortT* __restrict__ W,
    const float* __restrict__ bias, float* __restrict__ Cf, ushortT* __restrict__ Cb,
    int M, int N, int K) {
  int lane = threadIdx.x;
  int n0 = blockIdx.x * 64;
  int m0 = blockIdx.y * 16;
  int wr = lane & 15;
  size_t ar = (size_t)((m0 + wr < M) ? (m0 + wr) : (M - 1));
  v8f acc0 = {0,0,0,0,0,0,0,0}, acc1 = acc0, acc2 = acc0, acc3 = acc0;
  for (int k0 = 0; k0 < K; k0 += 32) {
    v16bf a  = load_frag(A, ar, K, k0, lane);
    v16bf b0 = load_frag(W, (size_t)(n0 +  0 + wr), K, k0, lane);
    v16bf b1 = load_frag(W, (size_t)(n0 + 16 + wr), K, k0, lane);
    v16bf b2 = load_frag(W, (size_t)(n0 + 32 + wr), K, k0, lane);
    v16bf b3 = load_frag(W, (size_t)(n0 + 48 + wr), K, k0, lane);
    acc0 = WMMA_BF16(a, b0, acc0);
    acc1 = WMMA_BF16(a, b1, acc1);
    acc2 = WMMA_BF16(a, b2, acc2);
    acc3 = WMMA_BF16(a, b3, acc3);
  }
  int hf = (lane >> 4) << 3;                     // +0 or +8 row offset
  #pragma unroll
  for (int r = 0; r < 8; ++r) {
    int row = m0 + r + hf;
    if (row >= M) continue;
    #pragma unroll
    for (int j = 0; j < 4; ++j) {
      float v = (j == 0) ? acc0[r] : (j == 1) ? acc1[r] : (j == 2) ? acc2[r] : acc3[r];
      int col = n0 + j * 16 + wr;
      if (bias) v += bias[col];
      size_t o = (size_t)row * N + col;
      if (Cf) Cf[o] = v;
      if (Cb) Cb[o] = f2bf(v);
    }
  }
}

// ---------------- fused BiGRU-direction scan with running time-max ----------------
// X: bf16 [B*T, Din] (row = b*T + t); Wcat: bf16 [1536, Din+512]
// outmax: f32 [B, 1024], this kernel writes cols [outColOff, outColOff+512)
__global__ __launch_bounds__(256) void k_gru_fused(
    const ushortT* __restrict__ X, int Din,
    const ushortT* __restrict__ Wcat,
    const float* __restrict__ bih, const float* __restrict__ bhh,
    float* __restrict__ outmax, int B, int T, int reverse, int outColOff) {
  extern __shared__ char smem[];
  ushortT* hs = (ushortT*)smem;                                   // [16][512]  bf16
  ushortT* Gx = (ushortT*)(smem + 16 * GH * 2);                   // [16][1536] bf16
  float*   Gh = (float*)  (smem + 16 * GH * 2 + 16 * THREE_H * 2);// [16][1536] f32
  const int Kcat = Din + GH;
  int tid = threadIdx.x, lane = tid & 31, wave = tid >> 5;
  int wr = lane & 15, hf = (lane >> 4) << 3;
  int b0 = blockIdx.x * 16;
  int brow = b0 + wr;
  size_t browc = (size_t)((brow < B) ? brow : (B - 1));
  for (int i = tid; i < 16 * GH; i += 256) hs[i] = 0;
  float mx[32];
  #pragma unroll
  for (int i = 0; i < 32; ++i) mx[i] = -3.0e38f;
  __syncthreads();

  for (int step = 0; step < T; ++step) {
    int t = reverse ? (T - 1 - step) : step;
    size_t xrow = browc * T + t;
    // --- WMMA phase: Gx = x_t @ Wih^T, Gh = h @ Whh^T (8 waves x 12 tiles = 96 N-tiles)
    for (int j = 0; j < 12; ++j) {
      int n0 = (wave * 12 + j) * 16;
      v8f ax = {0,0,0,0,0,0,0,0};
      for (int k0 = 0; k0 < Din; k0 += 32) {
        v16bf a = load_frag(X, xrow, Din, k0, lane);
        v16bf b = load_frag(Wcat, (size_t)(n0 + wr), Kcat, k0, lane);
        ax = WMMA_BF16(a, b, ax);
      }
      v8f ah = {0,0,0,0,0,0,0,0};
      #pragma unroll
      for (int k0 = 0; k0 < GH; k0 += 32) {
        v16bf a = load_frag(hs, (size_t)wr, GH, k0, lane);
        v16bf b = load_frag(Wcat, (size_t)(n0 + wr), Kcat, Din + k0, lane);
        ah = WMMA_BF16(a, b, ah);
      }
      #pragma unroll
      for (int r = 0; r < 8; ++r) {
        Gx[(r + hf) * THREE_H + n0 + wr] = f2bf(ax[r]);
        Gh[(r + hf) * THREE_H + n0 + wr] = ah[r];
      }
    }
    __syncthreads();
    // --- gate phase: each thread owns 32 fixed (row,col) cells
    #pragma unroll
    for (int i = 0; i < 32; ++i) {
      int idx = i * 256 + tid;
      int row = idx >> 9, col = idx & (GH - 1);
      float xr = bf2f(Gx[row * THREE_H + col])          + bih[col];
      float xz = bf2f(Gx[row * THREE_H + col + GH])     + bih[col + GH];
      float xn = bf2f(Gx[row * THREE_H + col + 2 * GH]) + bih[col + 2 * GH];
      float hr = Gh[row * THREE_H + col]          + bhh[col];
      float hz = Gh[row * THREE_H + col + GH]     + bhh[col + GH];
      float hn = Gh[row * THREE_H + col + 2 * GH] + bhh[col + 2 * GH];
      float rg = sigmoidf_(xr + hr);
      float zg = sigmoidf_(xz + hz);
      float ng = tanhf(xn + rg * hn);
      float hp = bf2f(hs[row * GH + col]);
      float hv = (1.0f - zg) * ng + zg * hp;
      mx[i] = fmaxf(mx[i], hv);
      hs[row * GH + col] = f2bf(hv);
    }
    __syncthreads();
  }
  #pragma unroll
  for (int i = 0; i < 32; ++i) {
    int idx = i * 256 + tid;
    int row = idx >> 9, col = idx & (GH - 1);
    int br = b0 + row;
    if (br < B) outmax[(size_t)br * TWO_H + outColOff + col] = mx[i];
  }
}

// ---------------- event attention (E=10) ----------------
__global__ __launch_bounds__(256) void k_attn(const float* __restrict__ M1,
    const float* __restrict__ EV, const float* __restrict__ esc,
    const float* __restrict__ relb, float* __restrict__ ectx) {
  __shared__ float red[256];
  __shared__ float sim[EPD];
  __shared__ float wgt[EPD];
  int blk = blockIdx.x, tid = threadIdx.x;
  int d = blk / SPD;
  const float* m  = M1 + (size_t)blk * TWO_H;
  const float* ev = EV + (size_t)d * EPD * TWO_H;
  for (int k = 0; k < EPD; ++k) {
    float p = 0.0f;
    for (int c = tid; c < TWO_H; c += 256) p += m[c] * ev[k * TWO_H + c];
    red[tid] = p; __syncthreads();
    for (int s = 128; s > 0; s >>= 1) { if (tid < s) red[tid] += red[tid + s]; __syncthreads(); }
    if (tid == 0) sim[k] = esc[d * EPD + k] * (red[0] + relb[0]);
    __syncthreads();
  }
  if (tid == 0) {
    float mxv = -3.0e38f;
    for (int k = 0; k < EPD; ++k) mxv = fmaxf(mxv, sim[k]);
    float s = 0.0f;
    for (int k = 0; k < EPD; ++k) { wgt[k] = __expf(sim[k] - mxv); s += wgt[k]; }
    for (int k = 0; k < EPD; ++k) wgt[k] /= s;
  }
  __syncthreads();
  for (int c = tid; c < TWO_H; c += 256) {
    float a = 0.0f;
    for (int k = 0; k < EPD; ++k) a += wgt[k] * ev[k * TWO_H + c];
    ectx[(size_t)blk * TWO_H + c] = a;
  }
}

// ---------------- small scoring kernels ----------------
__global__ void k_u(const float* __restrict__ W, const float* __restrict__ blog, float* __restrict__ u) {
  int e = blockIdx.x * blockDim.x + threadIdx.x;
  if (e >= TWO_H) return;
  float a = 0.0f;
  for (int f = 0; f < TWO_H; ++f) a += W[(size_t)e * TWO_H + f] * blog[f];
  u[e] = a;
}

__global__ __launch_bounds__(256) void k_doc_probs(const float* __restrict__ dv,
    const float* __restrict__ cw, const float* __restrict__ u, const float* __restrict__ dpe,
    const float* __restrict__ dpw, const float* __restrict__ dbias, float* __restrict__ out) {
  __shared__ float red[256];
  int d = blockIdx.x, tid = threadIdx.x;
  const float* v = dv + (size_t)d * TWO_H;
  float p = 0.0f;
  for (int c = tid; c < TWO_H; c += 256) p += v[c] * (cw[c] + u[c]);
  red[tid] = p; __syncthreads();
  for (int s = 128; s > 0; s >>= 1) { if (tid < s) red[tid] += red[tid + s]; __syncthreads(); }
  if (tid == 0) {
    int di = d * PDOC / NDOC;
    float pos = 0.0f;
    for (int i = 0; i < PDIM; ++i) pos += dpe[di * PDIM + i] * dpw[i];
    out[d] = red[0] + pos + dbias[0];
  }
}

__global__ __launch_bounds__(256) void k_sent_probs(const float* __restrict__ cur,
    const float* __restrict__ V, const float* __restrict__ cw, const float* __restrict__ dpe,
    const float* __restrict__ sdpw, const float* __restrict__ spe, const float* __restrict__ spw,
    const float* __restrict__ sbias, float* __restrict__ out) {
  __shared__ float red[256];
  int blk = blockIdx.x, tid = threadIdx.x;
  int d = blk / SPD, s = blk % SPD;
  const float* c  = cur + (size_t)blk * TWO_H;
  const float* vd = V + (size_t)d * TWO_H;
  float p = 0.0f;
  for (int i = tid; i < TWO_H; i += 256) p += c[i] * (cw[i] + vd[i]);
  red[tid] = p; __syncthreads();
  for (int t = 128; t > 0; t >>= 1) { if (tid < t) red[tid] += red[tid + t]; __syncthreads(); }
  if (tid == 0) {
    int di = d * PDOC / NDOC;
    int si = s * PSENT / SPD;
    float sdp = 0.0f, sp = 0.0f;
    for (int i = 0; i < PDIM; ++i) {
      sdp += dpe[di * PDIM + i] * sdpw[i];
      sp  += spe[si * PDIM + i] * spw[i];
    }
    out[blk] = red[0] + sdp + sp + sbias[0];
  }
}

// ---------------- host launcher ----------------
extern "C" void kernel_launch(void* const* d_in, const int* in_sizes, int n_in,
                              void* d_out, int out_size, void* d_ws, size_t ws_size,
                              hipStream_t stream) {
  (void)in_sizes; (void)n_in; (void)out_size; (void)ws_size;
  const int*   x      = (const int*)d_in[0];
  const int*   events = (const int*)d_in[1];
  const float* esc    = (const float*)d_in[2];
  auto F = [&](int i) { return (const float*)d_in[i]; };
  const float* embed  = F(5);
  float* out = (float*)d_out;

  // ---- workspace carve-out (256B aligned) ----
  char* ws = (char*)d_ws;
  size_t off = 0;
  auto take = [&](size_t bytes) { char* p = ws + off; off += (bytes + 255) & ~(size_t)255; return p; };
  const int KW = DPAD + GH;     // 832
  const int KS = TWO_H + GH;    // 1536
  ushortT* wcatWF = (ushortT*)take((size_t)THREE_H * KW * 2);
  ushortT* wcatWB = (ushortT*)take((size_t)THREE_H * KW * 2);
  ushortT* wcatSF = (ushortT*)take((size_t)THREE_H * KS * 2);
  ushortT* wcatSB = (ushortT*)take((size_t)THREE_H * KS * 2);
  ushortT* wcatDF = (ushortT*)take((size_t)THREE_H * KS * 2);
  ushortT* wcatDB = (ushortT*)take((size_t)THREE_H * KS * 2);
  ushortT* wcatEF = (ushortT*)take((size_t)THREE_H * KW * 2);
  ushortT* wcatEB = (ushortT*)take((size_t)THREE_H * KW * 2);
  ushortT* relT   = (ushortT*)take((size_t)TWO_H * TWO_H * 2);
  ushortT* preW   = (ushortT*)take((size_t)TWO_H * 2 * TWO_H * 2);
  ushortT* salW   = (ushortT*)take((size_t)TWO_H * 2 * TWO_H * 2);
  ushortT* embx   = (ushortT*)take((size_t)NDOC * SPD * LW * DPAD * 2);
  ushortT* embev  = (ushortT*)take((size_t)NDOC * EPD * EVL * DPAD * 2);
  float*   sentv  = (float*)  take((size_t)NDOC * SPD * TWO_H * 4);
  ushortT* sentbf = (ushortT*)take((size_t)NDOC * SPD * TWO_H * 2);
  float*   docv   = (float*)  take((size_t)NDOC * TWO_H * 4);
  ushortT* docbf  = (ushortT*)take((size_t)NDOC * TWO_H * 2);
  float*   blog   = (float*)  take((size_t)TWO_H * 4);
  float*   evv    = (float*)  take((size_t)NDOC * EPD * TWO_H * 4);
  float*   M1     = (float*)  take((size_t)NDOC * SPD * TWO_H * 4);
  float*   ectx   = (float*)  take((size_t)NDOC * SPD * TWO_H * 4);
  ushortT* xcur   = (ushortT*)take((size_t)NDOC * SPD * 2 * TWO_H * 2);
  float*   cur    = (float*)  take((size_t)NDOC * SPD * TWO_H * 4);
  ushortT* ctxbf  = (ushortT*)take((size_t)NDOC * 2 * TWO_H * 2);
  float*   Vbuf   = (float*)  take((size_t)NDOC * TWO_H * 4);
  float*   ubuf   = (float*)  take((size_t)TWO_H * 4);

  auto blocks = [](long n) { return (unsigned)((n + 255) / 256); };
  const size_t SCAN_SMEM = (size_t)16 * GH * 2 + (size_t)16 * THREE_H * 2 + (size_t)16 * THREE_H * 4;

  // ---- weight prep (fp32 -> bf16, concat [Wih|pad|Whh]) ----
  k_build_wcat<<<blocks((long)THREE_H * KW), 256, 0, stream>>>(wcatWF, F(6),  F(7),  DEMB, DPAD);
  k_build_wcat<<<blocks((long)THREE_H * KW), 256, 0, stream>>>(wcatWB, F(10), F(11), DEMB, DPAD);
  k_build_wcat<<<blocks((long)THREE_H * KS), 256, 0, stream>>>(wcatSF, F(14), F(15), TWO_H, TWO_H);
  k_build_wcat<<<blocks((long)THREE_H * KS), 256, 0, stream>>>(wcatSB, F(18), F(19), TWO_H, TWO_H);
  k_build_wcat<<<blocks((long)THREE_H * KS), 256, 0, stream>>>(wcatDF, F(22), F(23), TWO_H, TWO_H);
  k_build_wcat<<<blocks((long)THREE_H * KS), 256, 0, stream>>>(wcatDB, F(26), F(27), TWO_H, TWO_H);
  k_build_wcat<<<blocks((long)THREE_H * KW), 256, 0, stream>>>(wcatEF, F(30), F(31), DEMB, DPAD);
  k_build_wcat<<<blocks((long)THREE_H * KW), 256, 0, stream>>>(wcatEB, F(34), F(35), DEMB, DPAD);
  k_cvt_T  <<<blocks((long)TWO_H * TWO_H),     256, 0, stream>>>(relT, F(50), TWO_H, TWO_H);
  k_cvt_pad<<<blocks((long)TWO_H * 2 * TWO_H), 256, 0, stream>>>(preW, F(43), TWO_H, 2 * TWO_H, 2 * TWO_H);
  k_cvt_pad<<<blocks((long)TWO_H * 2 * TWO_H), 256, 0, stream>>>(salW, F(46), TWO_H, 2 * TWO_H, 2 * TWO_H);

  // ---- embedding gathers (pad D 300 -> 320, zero-filled) ----
  const long NTOK = (long)NDOC * SPD * LW;      // 40000
  const long NETK = (long)NDOC * EPD * EVL;     // 1600
  k_gather<<<blocks(NTOK * DPAD), 256, 0, stream>>>(embx,  x,      embed, NTOK);
  k_gather<<<blocks(NETK * DPAD), 256, 0, stream>>>(embev, events, embed, NETK);

  // ---- word-level BiGRU (B=1000, T=40) -> sent_vec [1000,1024] ----
  k_gru_fused<<<63, 256, SCAN_SMEM, stream>>>(embx, DPAD, wcatWF, F(8),  F(9),  sentv, 1000, LW, 0, 0);
  k_gru_fused<<<63, 256, SCAN_SMEM, stream>>>(embx, DPAD, wcatWB, F(12), F(13), sentv, 1000, LW, 1, GH);
  k_cvt_pad<<<blocks((long)1000 * TWO_H), 256, 0, stream>>>(sentbf, sentv, 1000, TWO_H, TWO_H);

  // ---- sentence-level BiGRU (B=40, T=25) -> doc_vec [40,1024] ----
  k_gru_fused<<<3, 256, SCAN_SMEM, stream>>>(sentbf, TWO_H, wcatSF, F(16), F(17), docv, NDOC, SPD, 0, 0);
  k_gru_fused<<<3, 256, SCAN_SMEM, stream>>>(sentbf, TWO_H, wcatSB, F(20), F(21), docv, NDOC, SPD, 1, GH);
  k_cvt_pad<<<blocks((long)NDOC * TWO_H), 256, 0, stream>>>(docbf, docv, NDOC, TWO_H, TWO_H);

  // ---- doc-level BiGRU (B=1, T=40) -> blog_vec [1024] ----
  k_gru_fused<<<1, 256, SCAN_SMEM, stream>>>(docbf, TWO_H, wcatDF, F(24), F(25), blog, 1, NDOC, 0, 0);
  k_gru_fused<<<1, 256, SCAN_SMEM, stream>>>(docbf, TWO_H, wcatDB, F(28), F(29), blog, 1, NDOC, 1, GH);

  // ---- event BiGRU (B=400, T=4) -> event_vec [400,1024] ----
  k_gru_fused<<<25, 256, SCAN_SMEM, stream>>>(embev, DPAD, wcatEF, F(32), F(33), evv, 400, EVL, 0, 0);
  k_gru_fused<<<25, 256, SCAN_SMEM, stream>>>(embev, DPAD, wcatEB, F(36), F(37), evv, 400, EVL, 1, GH);

  // ---- attention + sentence features ----
  k_gemm_bf16<<<dim3(TWO_H / 64, (1000 + 15) / 16), 32, 0, stream>>>(
      sentbf, relT, nullptr, M1, nullptr, 1000, TWO_H, TWO_H);
  k_attn<<<1000, 256, 0, stream>>>(M1, evv, esc, F(51), ectx);
  k_concat_bf16<<<blocks((long)1000 * 2 * TWO_H), 256, 0, stream>>>(xcur, sentv, ectx, 1000, TWO_H, TWO_H, 0);
  k_gemm_bf16<<<dim3(TWO_H / 64, (1000 + 15) / 16), 32, 0, stream>>>(
      xcur, preW, F(44), cur, nullptr, 1000, TWO_H, 2 * TWO_H);
  k_concat_bf16<<<blocks((long)NDOC * 2 * TWO_H), 256, 0, stream>>>(ctxbf, blog, docv, NDOC, TWO_H, TWO_H, 1);
  k_gemm_bf16<<<dim3(TWO_H / 64, (NDOC + 15) / 16), 32, 0, stream>>>(
      ctxbf, salW, nullptr, Vbuf, nullptr, NDOC, TWO_H, 2 * TWO_H);

  // ---- scores ----
  k_u<<<4, 256, 0, stream>>>(F(39), blog, ubuf);
  k_doc_probs<<<NDOC, 256, 0, stream>>>(docv, F(38), ubuf, F(40), F(41), F(42), out + 1000);
  k_sent_probs<<<1000, 256, 0, stream>>>(cur, Vbuf, F(45), F(40), F(47), F(48), F(49), F(53), out);
}
